// DecoderLayer_36490042147263
// MI455X (gfx1250) — matrix operand order repaired and checked by reading, not compiled, not run on hardware
//
#include <hip/hip_runtime.h>
#include <hip/hip_bf16.h>
#include <math.h>

typedef unsigned short u16;
typedef unsigned int   u32;
typedef __attribute__((ext_vector_type(16))) __bf16 v16bf;
typedef __attribute__((ext_vector_type(8)))  float  v8f;
typedef __attribute__((ext_vector_type(4)))  unsigned int v4u;
typedef __attribute__((ext_vector_type(8)))  int    v8i;
typedef __attribute__((ext_vector_type(4)))  int    v4i;

#if defined(__gfx1250__) && __has_builtin(__builtin_amdgcn_tensor_load_to_lds)
#define USE_TDM 1
#else
#define USE_TDM 0
#endif

// ---------- bf16 helpers (RNE) ----------
__device__ __forceinline__ float bf2f(u16 h) {
  u32 u = ((u32)h) << 16; float f; __builtin_memcpy(&f, &u, 4); return f;
}
__device__ __forceinline__ u16 f2bf(float f) {
  u32 u; __builtin_memcpy(&u, &f, 4);
  u += 0x7fffu + ((u >> 16) & 1u);
  return (u16)(u >> 16);
}

// ---------- converters ----------
__global__ void cvt_f2bf_kernel(const float* __restrict__ in, u16* __restrict__ out, size_t n) {
  size_t i = (size_t)blockIdx.x * blockDim.x + threadIdx.x;
  if (i < n) out[i] = f2bf(in[i]);
}
// in[R][C] (row-major) -> out[C][R]  (weight transpose to [K,N] B layout)
__global__ void cvt_f2bf_T_kernel(const float* __restrict__ in, u16* __restrict__ out, int R, int C) {
  size_t i = (size_t)blockIdx.x * blockDim.x + threadIdx.x;
  size_t n = (size_t)R * C;
  if (i >= n) return;
  int c = (int)(i % C), r = (int)(i / C);
  out[(size_t)c * R + r] = f2bf(in[i]);
}

// ---------- WMMA GEMM: C[M,N] = A[M,K](bf16) * B[K,N](bf16) + bias, opt GELU ----------
#define BM 128
#define BN 128
#define BK 32
#define LDA_U 17   // uints per LDS A row (34 bf16: 32 data + 1-dword TDM pad)
#define LDB_U 17   // uints per LDS Bt row (34 bf16, padded)

template<int ACT, int OBF>
__global__ __launch_bounds__(256) void gemm_bf16_wmma(
    const u16* __restrict__ A, const u16* __restrict__ Bm,
    const float* __restrict__ bias, void* __restrict__ Cout,
    int M, int N, int K)
{
  __shared__ u32 sA [BM * LDA_U];   // A tile, row-major [128][32] bf16 (+pad)
  __shared__ u32 sBt[BN * LDB_U];   // B tile transposed: [n][k] bf16
  u16* sBt16 = (u16*)sBt;

  const int tid  = threadIdx.x;
  const int ln   = tid & 31;
  const int wv   = tid >> 5;
  const int wrow = wv >> 2;       // 0..1  -> 64-row slab
  const int wcol = wv & 3;        // 0..3  -> 32-col slab
  const int lane15   = ln & 15;
  const int laneHalf = ln >> 4;   // 0/1

  const int bm0 = blockIdx.y * BM;
  const int bn0 = blockIdx.x * BN;

  // tile-loader mappings (256 threads)
  const int arow = tid >> 1;            // 0..127
  const int aseg = (tid & 1) * 8;       // uint offset in row: 0 or 8
  const int brow = tid >> 3;            // 0..31  (k within tile)
  const int bseg = (tid & 7) * 16;      // bf16 col offset: 0,16,...,112

#if USE_TDM
  // Uniform pieces of the Tensor DMA descriptor (D#) for the A tile.
  const unsigned lds_a = (unsigned)(size_t)(void*)sA;             // LDS byte offset
  const unsigned long long gbaseA =
      (unsigned long long)(size_t)(const void*)(A + (size_t)bm0 * K);
#endif

  v8f acc[4][2];
  {
    v8f z = {0.f,0.f,0.f,0.f,0.f,0.f,0.f,0.f};
    for (int mt = 0; mt < 4; ++mt)
      for (int nt = 0; nt < 2; ++nt) acc[mt][nt] = z;
  }

  for (int kb = 0; kb < K; kb += BK) {
#if USE_TDM
    // ---- TDM: DMA the 128x32 bf16 A tile into LDS (wave 0 only issues) ----
    if (tid < 32) {
      unsigned long long ga = gbaseA + (unsigned long long)kb * 2ull;  // byte addr of tile start
      v4u g0;
      g0.x = 1u;                                            // count=1, load, no gather
      g0.y = lds_a;                                         // lds_addr
      g0.z = (u32)ga;                                       // global_addr[31:0]
      g0.w = (u32)((ga >> 32) & 0x01FFFFFFull) | (2u << 30);// global_addr[56:32] | type=2
      v8i g1;
      // data_size=2B | pad_enable | pad_interval=16dw | pad_amount=1dw
      g1[0] = (int)((1u << 16) | (1u << 20) | (3u << 22));
      g1[1] = (int)(((u32)K & 0xFFFFu) << 16);              // tensor_dim0[15:0]
      g1[2] = (int)(((u32)K >> 16) | (((u32)M & 0xFFFFu) << 16)); // tensor_dim0[31:16] | tensor_dim1[15:0]
      g1[3] = (int)(((u32)M >> 16) | ((u32)BK << 16));      // tensor_dim1[31:16] | tile_dim0=32
      g1[4] = BM;                                           // tile_dim1=128 (tile_dim2=0)
      g1[5] = K;                                            // tensor_dim0_stride[31:0]
      g1[6] = 0; g1[7] = 0;
      v4i gz = {0, 0, 0, 0};
#if __clang_major__ >= 23
      v8i z8 = {0, 0, 0, 0, 0, 0, 0, 0};
      __builtin_amdgcn_tensor_load_to_lds(g0, g1, gz, gz, z8, 0);
#else
      __builtin_amdgcn_tensor_load_to_lds(g0, g1, gz, gz, 0);
#endif
    }
#else
    // ---- stage A tile (vector global loads, dword LDS stores) ----
    {
      const u16* gp = A + (size_t)(bm0 + arow) * K + kb + (tid & 1) * 16;
      uint4 p0 = *(const uint4*)gp;
      uint4 p1 = *(const uint4*)(gp + 8);
      u32* d = &sA[arow * LDA_U + aseg];
      d[0]=p0.x; d[1]=p0.y; d[2]=p0.z; d[3]=p0.w;
      d[4]=p1.x; d[5]=p1.y; d[6]=p1.z; d[7]=p1.w;
      if (kb + BK < K) __builtin_prefetch(gp + BK, 0, 0);
    }
#endif
    // ---- stage B tile transposed: sBt16[n*34 + k] = B[kb+k][bn0+n] ----
    {
      const u16* gp = Bm + (size_t)(kb + brow) * N + bn0 + bseg;
      union { uint4 q[2]; u16 h[16]; } tb;
      tb.q[0] = *(const uint4*)gp;
      tb.q[1] = *(const uint4*)(gp + 8);
      #pragma unroll
      for (int j = 0; j < 16; ++j)
        sBt16[(bseg + j) * (2 * LDB_U) + brow] = tb.h[j];
      if (kb + BK < K) __builtin_prefetch(gp + (size_t)BK * N, 0, 0);
    }
#if USE_TDM
    if (tid < 32) __builtin_amdgcn_s_wait_tensorcnt(0);   // wave 0: A tile landed in LDS
#endif
    __syncthreads();

    // ---- build fragments per CDNA5 WMMA layouts ----
    union FA { v16bf v; u32 u[8]; };
    FA a[4], b[2];
    #pragma unroll
    for (int mt = 0; mt < 4; ++mt) {
      // A 16x32: lane<16 rows M=lane K[0..7]+K[16..23]; lane>=16 K[8..15]+K[24..31]
      int r = wrow * 64 + mt * 16 + lane15;
      const u32* base = &sA[r * LDA_U + laneHalf * 4];
      #pragma unroll
      for (int j = 0; j < 4; ++j) { a[mt].u[j] = base[j]; a[mt].u[4 + j] = base[8 + j]; }
    }
    #pragma unroll
    for (int nt = 0; nt < 2; ++nt) {
      // B 32x16: lanes 0-15 N=lane K[0..7]+K[16..23]; lanes 16-31 K[8..15]+K[24..31]
      int n = wcol * 32 + nt * 16 + lane15;
      const u32* base = &sBt[n * LDB_U + laneHalf * 4];
      #pragma unroll
      for (int j = 0; j < 4; ++j) { b[nt].u[j] = base[j]; b[nt].u[4 + j] = base[8 + j]; }
    }

    #pragma unroll
    for (int mt = 0; mt < 4; ++mt)
      #pragma unroll
      for (int nt = 0; nt < 2; ++nt)
        acc[mt][nt] = __builtin_amdgcn_wmma_f32_16x16x32_bf16(
            false, a[mt].v, false, b[nt].v, (short)0, acc[mt][nt], false, false);

    __syncthreads();
  }

  // ---- epilogue: C 16x16 layout — VGPR i holds (row = laneHalf*8 + i, col = lane15) ----
  #pragma unroll
  for (int mt = 0; mt < 4; ++mt) {
    #pragma unroll
    for (int nt = 0; nt < 2; ++nt) {
      int gr0 = bm0 + wrow * 64 + mt * 16 + laneHalf * 8;
      int gc  = bn0 + wcol * 32 + nt * 16 + lane15;
      float bsv = bias ? bias[gc] : 0.f;
      #pragma unroll
      for (int i = 0; i < 8; ++i) {
        float val = acc[mt][nt][i] + bsv;
        if (ACT == 1) val = 0.5f * val * (1.f + erff(val * 0.70710678118654752f));
        size_t o = (size_t)(gr0 + i) * N + gc;
        if (OBF) ((u16*)Cout)[o]   = f2bf(val);
        else     ((float*)Cout)[o] = val;
      }
    }
  }
}

// ---------- autocorrelation: mean_corr[b,tau] = (1/D) sum_t sum_d q[t,d]*k[(t-tau)%L,d] ----------
__global__ __launch_bounds__(256) void corr_kernel(
    const u16* __restrict__ q, const u16* __restrict__ k,
    float* __restrict__ corr, int L, int D, int kStride)
{
  int b = blockIdx.y, tau = blockIdx.x, tid = threadIdx.x;
  const u16* qb = q + (size_t)b * L * D;
  const u16* kb = k + (size_t)b * kStride * D;
  float s = 0.f;
  for (int t = 0; t < L; ++t) {
    int tk = t - tau; if (tk < 0) tk += L;
    const u16* qr = qb + (size_t)t  * D;
    const u16* kr = kb + (size_t)tk * D;
    for (int d = tid; d < D; d += 256)
      s += bf2f(qr[d]) * bf2f(kr[d]);
  }
  __shared__ float red[256];
  red[tid] = s; __syncthreads();
  for (int w = 128; w > 0; w >>= 1) {
    if (tid < w) red[tid] += red[tid + w];
    __syncthreads();
  }
  if (tid == 0) corr[(size_t)b * L + tau] = red[0] / (float)D;
}

// ---------- top-5 + softmax per batch ----------
__global__ void topk_softmax_kernel(const float* __restrict__ corr,
                                    float* __restrict__ w, int* __restrict__ dly, int L)
{
  int b = blockIdx.x;
  if (threadIdx.x != 0) return;
  const float* c = corr + (size_t)b * L;
  float vals[5]; int idx[5];
  for (int k = 0; k < 5; ++k) {
    float best = -3.4e38f; int bi = 0;
    for (int t = 0; t < L; ++t) {
      bool used = false;
      for (int j = 0; j < k; ++j) used |= (idx[j] == t);
      if (!used && c[t] > best) { best = c[t]; bi = t; }
    }
    vals[k] = best; idx[k] = bi;
  }
  float mx = vals[0];
  for (int k = 1; k < 5; ++k) mx = fmaxf(mx, vals[k]);
  float se = 0.f;
  for (int k = 0; k < 5; ++k) { vals[k] = __expf(vals[k] - mx); se += vals[k]; }
  for (int k = 0; k < 5; ++k) { w[b * 5 + k] = vals[k] / se; dly[b * 5 + k] = idx[k]; }
}

// ---------- time-delay aggregation: out[b,l,d] = sum_k w[b,k]*v[b,(l+dly)%L,d] ----------
__global__ __launch_bounds__(256) void agg_kernel(
    const u16* __restrict__ v, const float* __restrict__ w, const int* __restrict__ dly,
    u16* __restrict__ out, int L, int D, int kStride, size_t total)
{
  size_t i = (size_t)blockIdx.x * blockDim.x + threadIdx.x;
  if (i >= total) return;
  int d = (int)(i % D);
  size_t bl = i / D;
  int l = (int)(bl % L);
  int b = (int)(bl / L);
  float s = 0.f;
  #pragma unroll
  for (int k = 0; k < 5; ++k) {
    int t = l + dly[b * 5 + k]; if (t >= L) t -= L;
    s += w[b * 5 + k] * bf2f(v[((size_t)b * kStride + t) * D + d]);
  }
  out[i] = f2bf(s);
}

// ---------- series_decomp of (inA + inB): seasonal + chained trend accumulation ----------
__global__ __launch_bounds__(256) void decomp_kernel(
    const float* __restrict__ inA, const float* __restrict__ inB,
    float* __restrict__ seasonal, const float* __restrict__ trend_in,
    float* __restrict__ trend_out, int L, int D, size_t total)
{
  size_t i = (size_t)blockIdx.x * blockDim.x + threadIdx.x;
  if (i >= total) return;
  int d = (int)(i % D);
  size_t bl = i / D;
  int l = (int)(bl % L);
  size_t brow = bl - (size_t)l;   // b*L
  float sum = 0.f;
  #pragma unroll
  for (int j = -12; j <= 12; ++j) {
    int lj = l + j; lj = lj < 0 ? 0 : (lj >= L ? L - 1 : lj);
    size_t o = (brow + (size_t)lj) * (size_t)D + d;
    sum += inA[o] + inB[o];
  }
  float tr  = sum * (1.f / 25.f);
  float cur = inA[i] + inB[i];
  seasonal[i]  = cur - tr;
  trend_out[i] = (trend_in ? trend_in[i] : 0.f) + tr;
}

// ======================================================================================
extern "C" void kernel_launch(void* const* d_in, const int* in_sizes, int n_in,
                              void* d_out, int out_size, void* d_ws, size_t ws_size,
                              hipStream_t stream) {
  (void)in_sizes; (void)n_in; (void)out_size; (void)ws_size;
  const int B = 16, L = 512, S = 1024, D = 1024, F = 4096;
  const size_t NL = (size_t)B * L * D;   // 8,388,608
  const size_t NS = (size_t)B * S * D;   // 16,777,216

  // ---- workspace carve ----
  char* ws = (char*)d_ws; size_t off = 0;
  auto carve = [&](size_t bytes) -> char* {
    char* p = ws + off; off += (bytes + 255) & ~(size_t)255; return p;
  };
  float* xA    = (float*)carve(NL * 4);
  float* xB    = (float*)carve(NL * 4);
  float* newx  = (float*)carve(NL * 4);
  float* trA   = (float*)carve(NL * 4);
  float* corrb = (float*)carve((size_t)B * L * 4);
  float* wts   = (float*)carve((size_t)B * 5 * 4);
  int*   dly   = (int*)  carve((size_t)B * 5 * 4);
  u16*   wbf   = (u16*)  carve((8 * (size_t)D * D + 2 * (size_t)D * F) * 2);
  u16*   bx    = (u16*)  carve(NS * 2);
  u16*   bq    = (u16*)  carve(NL * 2);
  u16*   bkv   = (u16*)  carve(2 * NS * 2);   // also aliased as FFN intermediate (NY == 2*NS)
  u16* bk = bkv;
  u16* bv = bkv + NS;
  u16* by = bkv;          // FFN intermediate aliases K/V (dead by then)
  u16* bagg = bq;         // agg output aliases Q (dead after corr)

  u16* w_sawq = wbf + 0 * (size_t)D * D;
  u16* w_sawk = wbf + 1 * (size_t)D * D;
  u16* w_sawv = wbf + 2 * (size_t)D * D;
  u16* w_sawo = wbf + 3 * (size_t)D * D;
  u16* w_cawq = wbf + 4 * (size_t)D * D;
  u16* w_cawk = wbf + 5 * (size_t)D * D;
  u16* w_cawv = wbf + 6 * (size_t)D * D;
  u16* w_cawo = wbf + 7 * (size_t)D * D;
  u16* w_c1   = wbf + 8 * (size_t)D * D;                    // [D][F]
  u16* w_c2   = w_c1 + (size_t)D * F;                       // [F][D]

  auto F32 = [&](int i) { return (const float*)d_in[i]; };
  auto blocks = [](size_t n) { return (unsigned)((n + 255) / 256); };

  // ---- weight conversions ----
  size_t DD = (size_t)D * D, DF = (size_t)D * F;
  cvt_f2bf_kernel<<<blocks(DD), 256, 0, stream>>>(F32(2),  w_sawq, DD);
  cvt_f2bf_kernel<<<blocks(DD), 256, 0, stream>>>(F32(4),  w_sawk, DD);
  cvt_f2bf_kernel<<<blocks(DD), 256, 0, stream>>>(F32(6),  w_sawv, DD);
  cvt_f2bf_kernel<<<blocks(DD), 256, 0, stream>>>(F32(8),  w_sawo, DD);
  cvt_f2bf_kernel<<<blocks(DD), 256, 0, stream>>>(F32(10), w_cawq, DD);
  cvt_f2bf_kernel<<<blocks(DD), 256, 0, stream>>>(F32(12), w_cawk, DD);
  cvt_f2bf_kernel<<<blocks(DD), 256, 0, stream>>>(F32(14), w_cawv, DD);
  cvt_f2bf_kernel<<<blocks(DD), 256, 0, stream>>>(F32(16), w_cawo, DD);
  cvt_f2bf_T_kernel<<<blocks(DF), 256, 0, stream>>>(F32(18), w_c1, F, D); // [F,D]->[D,F]
  cvt_f2bf_T_kernel<<<blocks(DF), 256, 0, stream>>>(F32(20), w_c2, D, F); // [D,F]->[F,D]

  auto gemm = [&](const u16* A, const u16* Bw, const float* bias, void* C,
                  int M, int N, int K, int act, int obf) {
    dim3 g(N / BN, M / BM);
    if (act == 0 && obf == 0) gemm_bf16_wmma<0,0><<<g, 256, 0, stream>>>(A, Bw, bias, C, M, N, K);
    else if (act == 0)        gemm_bf16_wmma<0,1><<<g, 256, 0, stream>>>(A, Bw, bias, C, M, N, K);
    else                      gemm_bf16_wmma<1,1><<<g, 256, 0, stream>>>(A, Bw, bias, C, M, N, K);
  };

  // ================= Phase 1: self-attention =================
  cvt_f2bf_kernel<<<blocks(NL), 256, 0, stream>>>(F32(0), bx, NL);
  gemm(bx, w_sawq, F32(3), bq, B * L, D, D, 0, 1);
  gemm(bx, w_sawk, F32(5), bk, B * L, D, D, 0, 1);
  gemm(bx, w_sawv, F32(7), bv, B * L, D, D, 0, 1);
  corr_kernel<<<dim3(L, B), 256, 0, stream>>>(bq, bk, corrb, L, D, L);
  topk_softmax_kernel<<<B, 32, 0, stream>>>(corrb, wts, dly, L);
  agg_kernel<<<blocks(NL), 256, 0, stream>>>(bv, wts, dly, bagg, L, D, L, NL);
  gemm(bagg, w_sawo, F32(9), newx, B * L, D, D, 0, 0);
  decomp_kernel<<<blocks(NL), 256, 0, stream>>>(F32(0), newx, xA, nullptr, trA, L, D, NL);

  // ================= Phase 2: cross-attention =================
  cvt_f2bf_kernel<<<blocks(NL), 256, 0, stream>>>(xA, bx, NL);
  gemm(bx, w_cawq, F32(11), bq, B * L, D, D, 0, 1);
  cvt_f2bf_kernel<<<blocks(NS), 256, 0, stream>>>(F32(1), bx, NS);
  gemm(bx, w_cawk, F32(13), bk, B * S, D, D, 0, 1);
  gemm(bx, w_cawv, F32(15), bv, B * S, D, D, 0, 1);
  corr_kernel<<<dim3(L, B), 256, 0, stream>>>(bq, bk, corrb, L, D, S);
  topk_softmax_kernel<<<B, 32, 0, stream>>>(corrb, wts, dly, L);
  agg_kernel<<<blocks(NL), 256, 0, stream>>>(bv, wts, dly, bagg, L, D, S, NL);
  gemm(bagg, w_cawo, F32(17), newx, B * L, D, D, 0, 0);
  decomp_kernel<<<blocks(NL), 256, 0, stream>>>(xA, newx, xB, trA, trA, L, D, NL);

  // ================= Phase 3: FFN =================
  cvt_f2bf_kernel<<<blocks(NL), 256, 0, stream>>>(xB, bx, NL);
  gemm(bx, w_c1, F32(19), by, B * L, F, D, 1, 1);        // conv1 + exact GELU -> bf16
  gemm(by, w_c2, F32(21), newx, B * L, D, F, 0, 0);      // conv2 -> fp32
  float* out_x  = (float*)d_out;
  float* out_tr = (float*)d_out + NL;
  decomp_kernel<<<blocks(NL), 256, 0, stream>>>(xB, newx, out_x, trA, out_tr, L, D, NL);
}
